// Net_40733469835603
// MI455X (gfx1250) — compile-verified, matrix operand order checked by reading, not run on hardware
//
#include <hip/hip_runtime.h>
#include <hip/hip_bf16.h>

// ---------------------------------------------------------------------------
// Types for CDNA5 WMMA
// ---------------------------------------------------------------------------
typedef __attribute__((ext_vector_type(16))) __bf16 v16bf;
typedef __attribute__((ext_vector_type(8)))  float  v8f;

__device__ __forceinline__ float bf2f(__bf16 b) {
    unsigned short u = __builtin_bit_cast(unsigned short, b);
    unsigned int x = ((unsigned int)u) << 16;
    return __builtin_bit_cast(float, x);
}
__device__ __forceinline__ __bf16 f2bf(float f) {
    unsigned int x = __builtin_bit_cast(unsigned int, f);
    unsigned int r = x + 0x7fffu + ((x >> 16) & 1u);   // round-to-nearest-even
    unsigned short u = (unsigned short)(r >> 16);
    return __builtin_bit_cast(__bf16, u);
}

static inline int cdiv(int a, int b) { return (a + b - 1) / b; }

// ---------------------------------------------------------------------------
// Fill kernels (d_ws is poisoned; we must init our own scratch)
// ---------------------------------------------------------------------------
__global__ void fill_f32_kernel(float* __restrict__ p, float v, int n) {
    int i = blockIdx.x * blockDim.x + threadIdx.x;
    if (i < n) p[i] = v;
}
__global__ void fill_u32_kernel(unsigned* __restrict__ p, unsigned v, int n) {
    int i = blockIdx.x * blockDim.x + threadIdx.x;
    if (i < n) p[i] = v;
}

// ---------------------------------------------------------------------------
// X conversion: f32 [N, Cin_real] -> bf16 [N, Kp], optional ones column
// ---------------------------------------------------------------------------
__global__ void convert_x_kernel(const float* __restrict__ x, __bf16* __restrict__ out,
                                 int N, int Cin_real, int add_ones, int Kp) {
    int i = blockIdx.x * blockDim.x + threadIdx.x;
    if (i >= N * Kp) return;
    int n = i / Kp, k = i - n * Kp;
    float v = 0.f;
    if (k < Cin_real)                 v = x[(size_t)n * Cin_real + k];
    else if (add_ones && k == Cin_real) v = 1.f;
    out[i] = f2bf(v);
}

// ---------------------------------------------------------------------------
// W conversion: W [125, Cin, Cout] + root [Cin, Cout] ->
// bf16 transposed B: out[nn * Kp + k], nn = kker*Cout + co  (kker==125 -> root)
// ---------------------------------------------------------------------------
__global__ void convert_w_kernel(const float* __restrict__ W, const float* __restrict__ root,
                                 __bf16* __restrict__ out, int Cin, int Cout, int Kp) {
    int total = 126 * Cout * Kp;
    int i = blockIdx.x * blockDim.x + threadIdx.x;
    if (i >= total) return;
    int nn = i / Kp, k = i - nn * Kp;
    int kk = nn / Cout, co = nn - kk * Cout;
    float v = 0.f;
    if (k < Cin)
        v = (kk < 125) ? W[((size_t)kk * Cin + k) * Cout + co]
                       : root[(size_t)k * Cout + co];
    out[i] = f2bf(v);
}

// ---------------------------------------------------------------------------
// bf16 WMMA GEMM:  C[M, Nn] = A[M, Kp] * B(Kp, Nn)   (B stored transposed [Nn, Kp])
// Block = 256 threads = 8 waves; block tile 128(M) x 64(N); wave tile 16 x 64.
// Kp multiple of 32, Nn multiple of 64.
// ---------------------------------------------------------------------------
__global__ __launch_bounds__(256) void gemm_bf16_wmma_kernel(
    const __bf16* __restrict__ A, const __bf16* __restrict__ BT,
    __bf16* __restrict__ C, int M, int Kp, int Nn) {
    const int wave = threadIdx.x >> 5;
    const int lane = threadIdx.x & 31;
    const int tiles_n = Nn >> 6;
    const int bm = blockIdx.x / tiles_n;
    const int bn = blockIdx.x - bm * tiles_n;
    const int row0 = bm * 128 + wave * 16;
    const int col0 = bn * 64;
    const int half = lane >> 4;
    const int l16  = lane & 15;
    int arow = row0 + l16;
    int arow_c = (arow < M) ? arow : (M - 1);   // clamp; extra rows never stored

    v8f acc0 = {}, acc1 = {}, acc2 = {}, acc3 = {};
    for (int k0 = 0; k0 < Kp; k0 += 32) {
        // A fragment: lane<16 -> K 0..15, lane>=16 -> K 16..31 of this step
        v16bf af = *(const v16bf*)(A + (size_t)arow_c * Kp + k0 + half * 16);
        // B fragments: column-contiguous thanks to transposed storage
        const __bf16* bb = BT + (size_t)(col0 + l16) * Kp + k0 + half * 16;
        v16bf b0 = *(const v16bf*)(bb + (size_t)0  * Kp * 16);
        v16bf b1 = *(const v16bf*)(bb + (size_t)16 * Kp);
        v16bf b2 = *(const v16bf*)(bb + (size_t)32 * Kp);
        v16bf b3 = *(const v16bf*)(bb + (size_t)48 * Kp);
        acc0 = __builtin_amdgcn_wmma_f32_16x16x32_bf16(false, af, false, b0, (short)0, acc0, false, false);
        acc1 = __builtin_amdgcn_wmma_f32_16x16x32_bf16(false, af, false, b1, (short)0, acc1, false, false);
        acc2 = __builtin_amdgcn_wmma_f32_16x16x32_bf16(false, af, false, b2, (short)0, acc2, false, false);
        acc3 = __builtin_amdgcn_wmma_f32_16x16x32_bf16(false, af, false, b3, (short)0, acc3, false, false);
    }
    // C/D layout: VGPR r holds row (r + 8*half), lanes 0..15 -> cols 0..15
    for (int r = 0; r < 8; ++r) {
        int row = row0 + r + 8 * half;
        if (row < M) {
            __bf16* cp = C + (size_t)row * Nn + col0 + l16;
            cp[0]  = f2bf(acc0[r]);
            cp[16] = f2bf(acc1[r]);
            cp[32] = f2bf(acc2[r]);
            cp[48] = f2bf(acc3[r]);
        }
    }
}

// ---------------------------------------------------------------------------
// Spline edge gather: one wave32 per edge, lane = output channel (mod 32).
// Z row stride is 126*Cout (slot 125 = root transform, untouched here).
// ---------------------------------------------------------------------------
__global__ __launch_bounds__(256) void spline_edge_kernel(
    const int* __restrict__ edges, const float* __restrict__ pseudo,
    const __bf16* __restrict__ Z, float* __restrict__ agg, float* __restrict__ deg,
    int E, int Cout) {
    int wid = (blockIdx.x * blockDim.x + threadIdx.x) >> 5;
    int lane = threadIdx.x & 31;
    if (wid >= E) return;
    int src = edges[wid];
    int dst = edges[E + wid];
    float p0 = pseudo[wid * 3 + 0] * 4.f;
    float p1 = pseudo[wid * 3 + 1] * 4.f;
    float p2 = pseudo[wid * 3 + 2] * 4.f;
    float i0 = fminf(fmaxf(floorf(p0), 0.f), 3.f);
    float i1 = fminf(fmaxf(floorf(p1), 0.f), 3.f);
    float i2 = fminf(fmaxf(floorf(p2), 0.f), 3.f);
    float f0 = p0 - i0, f1 = p1 - i1, f2 = p2 - i2;
    int kbase = ((int)i0 * 5 + (int)i1) * 5 + (int)i2;
    float wx[2] = {1.f - f0, f0}, wy[2] = {1.f - f1, f1}, wz[2] = {1.f - f2, f2};

    int nreg = Cout >> 5;                       // Cout in {32,64,96,128}
    float accum[4] = {0.f, 0.f, 0.f, 0.f};
    const size_t zrow = (size_t)src * 126;
    #pragma unroll
    for (int a = 0; a < 2; ++a)
      #pragma unroll
      for (int b = 0; b < 2; ++b)
        #pragma unroll
        for (int c = 0; c < 2; ++c) {
            int kk = kbase + a * 25 + b * 5 + c;
            float w = wx[a] * wy[b] * wz[c];
            const __bf16* zr = Z + (zrow + kk) * Cout + lane;
            for (int r = 0; r < nreg; ++r) accum[r] += w * bf2f(zr[r * 32]);
        }
    float* ad = agg + (size_t)dst * Cout + lane;
    for (int r = 0; r < nreg; ++r) atomicAdd(ad + r * 32, accum[r]);
    if (lane == 0) atomicAdd(deg + dst, 1.0f);
}

// ---------------------------------------------------------------------------
// Node update: out = agg/max(deg,1) + Z[n,125,:] + bias (+res) (+ELU)
// ---------------------------------------------------------------------------
__global__ void node_update_kernel(const float* __restrict__ agg, const float* __restrict__ deg,
                                   const __bf16* __restrict__ Z, const float* __restrict__ bias,
                                   const float* __restrict__ res, float* __restrict__ out,
                                   int N, int Cout, int apply_elu) {
    int i = blockIdx.x * blockDim.x + threadIdx.x;
    if (i >= N * Cout) return;
    int n = i / Cout, co = i - n * Cout;
    float d = deg[n]; d = d < 1.f ? 1.f : d;
    float v = agg[i] / d + bf2f(Z[((size_t)n * 126 + 125) * Cout + co]) + bias[co];
    if (res) v += res[i];
    if (apply_elu) v = v > 0.f ? v : (expf(v) - 1.f);
    out[i] = v;
}

// ---------------------------------------------------------------------------
// Small dense linear (skip connections): out[n,co] = x @ w + b
// ---------------------------------------------------------------------------
__global__ void linear_kernel(const float* __restrict__ x, const float* __restrict__ w,
                              const float* __restrict__ b, float* __restrict__ out,
                              int N, int Ci, int Co) {
    int i = blockIdx.x * blockDim.x + threadIdx.x;
    if (i >= N * Co) return;
    int n = i / Co, co = i - n * Co;
    float acc = b[co];
    for (int k = 0; k < Ci; ++k) acc += x[(size_t)n * Ci + k] * w[(size_t)k * Co + co];
    out[i] = acc;
}

// ---------------------------------------------------------------------------
// Segment max pool via order-preserving uint encoding + atomicMax
// ---------------------------------------------------------------------------
__device__ __forceinline__ unsigned encf(float f) {
    unsigned u = __float_as_uint(f);
    return (u & 0x80000000u) ? ~u : (u | 0x80000000u);
}
__device__ __forceinline__ float decf(unsigned e) {
    unsigned u = (e & 0x80000000u) ? (e ^ 0x80000000u) : ~e;
    return __uint_as_float(u);
}
#define POOL_NEG_INF_ENC 0x007FFFFFu   // encf(-inf)

__global__ void pool_scatter_kernel(const float* __restrict__ x, const int* __restrict__ cluster,
                                    unsigned* __restrict__ pool, int Nin, int C) {
    int i = blockIdx.x * blockDim.x + threadIdx.x;
    if (i >= Nin * C) return;
    int n = i / C, c = i - n * C;
    atomicMax(pool + (size_t)cluster[n] * C + c, encf(x[i]));
}
__global__ void pool_final_kernel(const unsigned* __restrict__ pool, float* __restrict__ out, int n) {
    int i = blockIdx.x * blockDim.x + threadIdx.x;
    if (i >= n) return;
    unsigned e = pool[i];
    out[i] = (e == POOL_NEG_INF_ENC) ? 0.f : decf(e);   // empty voxel -> 0
}

// ---------------------------------------------------------------------------
// Readout: scatter-mean, fc1, log-softmax
// ---------------------------------------------------------------------------
__global__ void scatter_mean_kernel(const float* __restrict__ x, const int* __restrict__ batch,
                                    float* __restrict__ gsum, float* __restrict__ gcnt,
                                    int N, int C) {
    int i = blockIdx.x * blockDim.x + threadIdx.x;
    if (i >= N * C) return;
    int n = i / C, c = i - n * C;
    atomicAdd(gsum + (size_t)batch[n] * C + c, x[i]);
    if (c == 0) atomicAdd(gcnt + batch[n], 1.0f);
}
__global__ void head_kernel(const float* __restrict__ gsum, const float* __restrict__ gcnt,
                            const float* __restrict__ fw, const float* __restrict__ fb,
                            float* __restrict__ out) {   // out [8,40]
    __shared__ float g[128];
    __shared__ float logits[40];
    __shared__ float s_lse;
    int b = blockIdx.x;
    int t = threadIdx.x;
    float cnt = gcnt[b]; cnt = cnt < 1.f ? 1.f : cnt;
    for (int c = t; c < 128; c += blockDim.x) g[c] = gsum[b * 128 + c] / cnt;
    __syncthreads();
    if (t < 40) {
        float a = fb[t];
        for (int k = 0; k < 128; ++k) a += g[k] * fw[k * 40 + t];
        logits[t] = a;
    }
    __syncthreads();
    if (t == 0) {
        float m = -3.4e38f;
        for (int k = 0; k < 40; ++k) m = fmaxf(m, logits[k]);
        float s = 0.f;
        for (int k = 0; k < 40; ++k) s += expf(logits[k] - m);
        s_lse = logf(s) + m;
    }
    __syncthreads();
    if (t < 40) out[b * 40 + t] = logits[t] - s_lse;
}

// ---------------------------------------------------------------------------
// Host orchestration
// ---------------------------------------------------------------------------
static const int cN0 = 20000, cN1 = 8000, cN2 = 2500, cN3 = 800, cN4 = 256;
static const int cE0 = 320000, cE1 = 128000, cE2 = 40000, cE3 = 12800, cE4 = 4096;

extern "C" void kernel_launch(void* const* d_in, const int* in_sizes, int n_in,
                              void* d_out, int out_size, void* d_ws, size_t ws_size,
                              hipStream_t stream) {
    (void)in_sizes; (void)n_in; (void)out_size; (void)ws_size;

    const float* x0  = (const float*)d_in[0];
    const int* eds[5]  = {(const int*)d_in[1], (const int*)d_in[3], (const int*)d_in[5],
                          (const int*)d_in[7], (const int*)d_in[9]};
    const float* pss[5] = {(const float*)d_in[2], (const float*)d_in[4], (const float*)d_in[6],
                           (const float*)d_in[8], (const float*)d_in[10]};
    const int* cl[4] = {(const int*)d_in[11], (const int*)d_in[12],
                        (const int*)d_in[13], (const int*)d_in[14]};
    const int* batch4 = (const int*)d_in[15];
    // conv param index c in _CONVS order -> w,root,b at 16+3c..18+3c
    auto CW = [&](int c) { return (const float*)d_in[16 + 3 * c]; };
    auto CR = [&](int c) { return (const float*)d_in[17 + 3 * c]; };
    auto CB = [&](int c) { return (const float*)d_in[18 + 3 * c]; };
    const float* skip1_w = (const float*)d_in[55]; const float* skip1_b = (const float*)d_in[56];
    const float* skip2_w = (const float*)d_in[57]; const float* skip2_b = (const float*)d_in[58];
    const float* skip3_w = (const float*)d_in[59]; const float* skip3_b = (const float*)d_in[60];
    const float* fc1_w   = (const float*)d_in[61]; const float* fc1_b   = (const float*)d_in[62];

    // ---- workspace carve-out ----
    char* ws = (char*)d_ws;
    size_t off = 0;
    auto take = [&](size_t bytes) -> void* {
        off = (off + 255) & ~(size_t)255;
        void* p = ws + off;
        off += bytes;
        return p;
    };
    __bf16*  Zb   = (__bf16*)take((size_t)cN0 * 126 * 32 * 2);  // max Z: conv1 (80.64M bf16)
    __bf16*  Xb   = (__bf16*)take((size_t)cN0 * 32 * 2);        // max padded X
    __bf16*  Wt   = (__bf16*)take((size_t)126 * 128 * 128 * 2); // max transposed W
    float*   agg  = (float*)take((size_t)cN0 * 32 * 4);
    float*   deg  = (float*)take((size_t)cN0 * 4);
    unsigned* plb = (unsigned*)take((size_t)cN1 * 32 * 4);      // max pool out elems
    float* f[6];
    for (int i = 0; i < 6; ++i) f[i] = (float*)take((size_t)cN0 * 32 * 4);
    float* gsum = (float*)take(8 * 128 * 4);
    float* gcnt = (float*)take(8 * 4);

    const int TPB = 256;
    auto fillf = [&](float* p, float v, int n) {
        fill_f32_kernel<<<cdiv(n, TPB), TPB, 0, stream>>>(p, v, n);
    };

    // ---- one spline conv: convert -> WMMA GEMM -> edge gather -> node update ----
    auto run_conv = [&](const float* xin, int N, int Cin_real, int add_ones, int Cout,
                        const int* edges, const float* pseudo, int E, int cidx,
                        const float* res, int elu, float* out) {
        int Cin = Cin_real + (add_ones ? 1 : 0);
        int Kp = ((Cin + 31) / 32) * 32;
        int Nn = 126 * Cout;
        convert_x_kernel<<<cdiv(N * Kp, TPB), TPB, 0, stream>>>(xin, Xb, N, Cin_real, add_ones, Kp);
        convert_w_kernel<<<cdiv(Nn * Kp, TPB), TPB, 0, stream>>>(CW(cidx), CR(cidx), Wt, Cin, Cout, Kp);
        fillf(agg, 0.f, N * Cout);
        fillf(deg, 0.f, N);
        int gblocks = cdiv(N, 128) * (Nn / 64);
        gemm_bf16_wmma_kernel<<<gblocks, TPB, 0, stream>>>(Xb, Wt, Zb, N, Kp, Nn);
        spline_edge_kernel<<<cdiv(E, 8), TPB, 0, stream>>>(edges, pseudo, Zb, agg, deg, E, Cout);
        node_update_kernel<<<cdiv(N * Cout, TPB), TPB, 0, stream>>>(agg, deg, Zb, CB(cidx),
                                                                    res, out, N, Cout, elu);
    };
    auto run_pool = [&](const float* xin, const int* cluster, int Nin, int Nout, int C, float* out) {
        fill_u32_kernel<<<cdiv(Nout * C, TPB), TPB, 0, stream>>>(plb, POOL_NEG_INF_ENC, Nout * C);
        pool_scatter_kernel<<<cdiv(Nin * C, TPB), TPB, 0, stream>>>(xin, cluster, plb, Nin, C);
        pool_final_kernel<<<cdiv(Nout * C, TPB), TPB, 0, stream>>>(plb, out, Nout * C);
    };

    // ---------------- network ----------------
    run_conv(x0, cN0, 1, 0, 32, eds[0], pss[0], cE0, 0, nullptr, 1, f[0]);      // t1 = elu(conv1)
    run_pool(f[0], cl[0], cN0, cN1, 32, f[1]);                                   // d1
    run_conv(f[1], cN1, 32, 1, 64, eds[1], pss[1], cE1, 1, nullptr, 1, f[2]);   // xa = elu(conv2)
    linear_kernel<<<cdiv(cN1 * 64, TPB), TPB, 0, stream>>>(f[1], skip1_w, skip1_b, f[3], cN1, 32, 64);
    run_conv(f[2], cN1, 64, 0, 64, eds[1], pss[1], cE1, 2, f[3], 1, f[4]);      // d2 = elu(conv22+s1)
    run_conv(f[4], cN1, 64, 0, 64, eds[1], pss[1], cE1, 3, nullptr, 1, f[2]);   // xa = elu(conv3)
    run_conv(f[2], cN1, 64, 0, 64, eds[1], pss[1], cE1, 4, f[4], 1, f[5]);      // d3 = elu(conv32+d2)
    run_pool(f[5], cl[1], cN1, cN2, 64, f[1]);                                   // d4
    run_conv(f[1], cN2, 64, 1, 96, eds[2], pss[2], cE2, 5, nullptr, 1, f[2]);   // xa = elu(conv4)
    linear_kernel<<<cdiv(cN2 * 96, TPB), TPB, 0, stream>>>(f[1], skip2_w, skip2_b, f[3], cN2, 64, 96);
    run_conv(f[2], cN2, 96, 0, 96, eds[2], pss[2], cE2, 6, f[3], 1, f[4]);      // d5 = elu(conv42+s2)
    run_pool(f[4], cl[2], cN2, cN3, 96, f[1]);                                   // d6
    run_conv(f[1], cN3, 96, 1, 96, eds[3], pss[3], cE3, 7, nullptr, 1, f[2]);   // xa = elu(conv5)
    run_conv(f[2], cN3, 96, 0, 96, eds[3], pss[3], cE3, 8, f[1], 1, f[4]);      // d7 = elu(conv52+d6)
    run_pool(f[4], cl[3], cN3, cN4, 96, f[1]);                                   // d8
    run_conv(f[1], cN4, 96, 1, 128, eds[4], pss[4], cE4, 9, nullptr, 1, f[2]);  // xa = elu(conv6)
    linear_kernel<<<cdiv(cN4 * 128, TPB), TPB, 0, stream>>>(f[1], skip3_w, skip3_b, f[3], cN4, 96, 128);
    run_conv(f[2], cN4, 128, 0, 128, eds[4], pss[4], cE4, 10, f[3], 1, f[4]);   // d9 = elu(conv62+s3)
    run_conv(f[4], cN4, 128, 0, 128, eds[4], pss[4], cE4, 11, nullptr, 1, f[2]);// xa = elu(conv7)
    run_conv(f[2], cN4, 128, 0, 128, eds[4], pss[4], cE4, 12, f[4], 1, f[5]);   // d10 = elu(conv72+d9)

    fillf(gsum, 0.f, 8 * 128);
    fillf(gcnt, 0.f, 8);
    scatter_mean_kernel<<<cdiv(cN4 * 128, TPB), TPB, 0, stream>>>(f[5], batch4, gsum, gcnt, cN4, 128);
    head_kernel<<<8, 64, 0, stream>>>(gsum, gcnt, fc1_w, fc1_b, (float*)d_out);
}
